// MultiHeadCrossAttention_26508538151323
// MI455X (gfx1250) — compile-verified, matrix operand order checked by reading, not run on hardware
//
#include <hip/hip_runtime.h>

// ---------------- problem constants ----------------
#define B_    16
#define S_    8192
#define E_    512
#define H_    8
#define NSB_  16      // S-blocks per batch
#define ROWS_ 512     // S rows per workgroup (S_/NSB_)
#define NPASS_ 16     // 32-row passes per workgroup (ROWS_/32)
#define KSTEPS 16     // 512 / 32 k-steps per WMMA chain

// ---------------- CDNA5 types ----------------
typedef _Float16     v16h  __attribute__((ext_vector_type(16)));
typedef float        v8f   __attribute__((ext_vector_type(8)));
typedef unsigned int u32x4 __attribute__((ext_vector_type(4)));
typedef int          i32x4 __attribute__((ext_vector_type(4)));
typedef int          i32x8 __attribute__((ext_vector_type(8)));

union Frag { unsigned u[8]; uint4 q[2]; v16h v; };

// v_cvt_pk_rtz_f16_f32: one VALU op packs two floats into 2 x f16
__device__ __forceinline__ unsigned pack2h(float a, float b) {
  auto pk = __builtin_amdgcn_cvt_pkrtz(a, b);
  unsigned r; __builtin_memcpy(&r, &pk, sizeof(r));
  return r;
}
// A-fragment K position for VGPR v, lane-half h2 (ISA 7.12.2, 16-bit A 16x32):
//   v<4 : K = h2*8 + 2v      v>=4 : K = 16 + h2*8 + 2(v-4)
__device__ __forceinline__ int kbase(int v, int h2) {
  return (v < 4) ? (h2 * 8 + 2 * v) : (16 + h2 * 8 + 2 * (v - 4));
}
// Build a 16x32 f16 A-fragment from a row-major fp32 row: each half needs
// 8+8 consecutive floats -> 4 x 16B loads + 8 x v_cvt_pk_rtz_f16_f32.
__device__ __forceinline__ Frag make_afrag_f32(const float* row, int kk, int h2) {
  const float4* p0 = (const float4*)(row + kk * 32 + h2 * 8);
  const float4* p1 = (const float4*)(row + kk * 32 + 16 + h2 * 8);
  float4 f0 = p0[0], f1 = p0[1], f2 = p1[0], f3 = p1[1];
  Frag a;
  a.u[0] = pack2h(f0.x, f0.y); a.u[1] = pack2h(f0.z, f0.w);
  a.u[2] = pack2h(f1.x, f1.y); a.u[3] = pack2h(f1.z, f1.w);
  a.u[4] = pack2h(f2.x, f2.y); a.u[5] = pack2h(f2.z, f2.w);
  a.u[6] = pack2h(f3.x, f3.y); a.u[7] = pack2h(f3.z, f3.w);
  return a;
}

// ---------------- Tensor Data Mover staging (32 x 512 fp32 tile -> LDS) ----
#if __has_builtin(__builtin_amdgcn_tensor_load_to_lds)
#define HAVE_TDM 1
__device__ __forceinline__ void tdm_load_tile(const float* g, unsigned lds_off) {
  unsigned long long ga = (unsigned long long)g;
  u32x4 g0;
  g0[0] = 1u;                                              // count=1 (user D#)
  g0[1] = lds_off;                                         // lds_addr (bytes)
  g0[2] = (unsigned)ga;                                    // global_addr[31:0]
  g0[3] = (unsigned)((ga >> 32) & 0x1FFFFFFu) | (2u << 30);// addr[56:32] | type=2
  i32x8 g1;
  g1[0] = (int)(2u << 16);    // data_size=4B, no multicast/pad/iterate
  g1[1] = (int)(512u << 16);  // tensor_dim0 lo16 = 512 (bits 63:48)
  g1[2] = (int)(32u << 16);   // tensor_dim1 lo16 = 32  (bits 95:80)
  g1[3] = (int)(512u << 16);  // tile_dim0 = 512        (bits 127:112)
  g1[4] = 32;                 // tile_dim1 = 32
  g1[5] = 512;                // tensor_dim0_stride lo32
  g1[6] = 0; g1[7] = 0;
  i32x4 g2 = (i32x4)0, g3 = (i32x4)0;
  i32x8 g4 = (i32x8)0;        // unused trailing group (6-arg builtin form)
  __builtin_amdgcn_tensor_load_to_lds(g0, g1, g2, g3, g4, 0);
}
#else
#define HAVE_TDM 0
#endif

// ---------------- kernel 1: repack weights fp32 -> f16 B-fragment layout ----
// Layout: dword index = ((nt*16 + kk)*32 + lane)*8 + v ; each dword = 2 f16
// holding W[k, c], W[k+1, c] with c = nt*16 + lane%16, k = kk*32 + kbase(v, lane/16).
__global__ __launch_bounds__(256) void pack_weights_kernel(
    const float* __restrict__ Wq, const float* __restrict__ Wk,
    const float* __restrict__ Wv, const float* __restrict__ Wo,
    unsigned* __restrict__ wqp, unsigned* __restrict__ wkp,
    unsigned* __restrict__ wvp, unsigned* __restrict__ wop) {
  unsigned idx = blockIdx.x * 256u + threadIdx.x;   // 4 * 131072 total
  unsigned m   = idx >> 17;
  unsigned rem = idx & 0x1FFFFu;
  const float* W = (m == 0) ? Wq : (m == 1) ? Wk : (m == 2) ? Wv : Wo;
  unsigned*  dst = (m == 0) ? wqp : (m == 1) ? wkp : (m == 2) ? wvp : wop;
  int v = rem & 7, lane = (rem >> 3) & 31, kk = (rem >> 8) & 15, nt = rem >> 12;
  int c = nt * 16 + (lane & 15);
  int k = kk * 32 + kbase(v, lane >> 4);
  dst[rem] = pack2h(W[(size_t)k * E_ + c], W[(size_t)(k + 1) * E_ + c]);
}

// ---------------- kernel 2: q = query @ Wq + bq  (M=16 batches) -------------
__global__ __launch_bounds__(256) void qproj_kernel(
    const float* __restrict__ query, const unsigned* __restrict__ wqp,
    const float* __restrict__ bq, float* __restrict__ qout) {
  int lane = threadIdx.x & 31, w = threadIdx.x >> 5;
  int h2 = lane >> 4, c16 = lane & 15;
  const uint4* wp4 = (const uint4*)wqp;
  v8f acc[4] = {};
  for (int kk = 0; kk < KSTEPS; ++kk) {
    Frag b[4];
#pragma unroll
    for (int t = 0; t < 4; ++t) {      // issue all B loads first (one clause)
      const uint4* p = wp4 + ((size_t)((w * 4 + t) * KSTEPS + kk) * 32 + lane) * 2;
      b[t].q[0] = p[0]; b[t].q[1] = p[1];
    }
    Frag a = make_afrag_f32(&query[c16 * E_], kk, h2);
#pragma unroll
    for (int t = 0; t < 4; ++t)        // back-to-back WMMA
      acc[t] = __builtin_amdgcn_wmma_f32_16x16x32_f16(
          false, a.v, false, b[t].v, (short)0, acc[t], false, false);
  }
#pragma unroll
  for (int t = 0; t < 4; ++t) {
    int c = (w * 4 + t) * 16 + c16;
    float bias = bq[c];
#pragma unroll
    for (int j = 0; j < 8; ++j) qout[(j + 8 * h2) * E_ + c] = acc[t][j] + bias;
  }
}

// ---------------- kernel 3: fused K/V projection + flash attention ----------
// grid = B_*NSB_ workgroups; 8 waves; wave w handles head h=w (its 64 E-cols).
// Each pass covers 32 s-rows (2 M-tiles) so every B-fragment load feeds 2 WMMAs.
__global__ __launch_bounds__(256) void attn_partial_kernel(
    const float* __restrict__ kv, const float* __restrict__ qws,
    const unsigned* __restrict__ wkp, const unsigned* __restrict__ wvp,
    const float* __restrict__ bk, const float* __restrict__ bv,
    float* __restrict__ part) {
  __shared__ float lds_kv[2][32 * E_];   // 2 x 64KB double buffer (TDM target)
  int b = blockIdx.x / NSB_, sb = blockIdx.x % NSB_;
  int lane = threadIdx.x & 31, h = threadIdx.x >> 5;
  int h2 = lane >> 4, c16 = lane & 15;
  const float* kvb = kv + ((size_t)b * S_ + (size_t)sb * ROWS_) * E_;
  const uint4* wk4 = (const uint4*)wkp;
  const uint4* wv4 = (const uint4*)wvp;

  float qv[4], bkv[4], bvv[4];
#pragma unroll
  for (int t = 0; t < 4; ++t) {
    int c = h * 64 + t * 16 + c16;
    qv[t] = qws[b * E_ + c]; bkv[t] = bk[c]; bvv[t] = bv[c];
  }
  float m = -1e30f, lsum = 0.f, ctx[4] = {0, 0, 0, 0};

#if HAVE_TDM
  if (h == 0)  // wave-uniform branch: TDM ignores EXEC, so gate at wave level
    tdm_load_tile(kvb, (unsigned)(unsigned long long)&lds_kv[0][0]);
#endif
  for (int pass = 0; pass < NPASS_; ++pass) {
    const float* cur = lds_kv[pass & 1];
#if HAVE_TDM
    if (h == 0) {
      if (pass + 1 < NPASS_) {
        tdm_load_tile(kvb + (size_t)(pass + 1) * 32 * E_,
                      (unsigned)(unsigned long long)&lds_kv[(pass + 1) & 1][0]);
        __builtin_amdgcn_s_wait_tensorcnt((short)1);   // current pass landed
      } else {
        __builtin_amdgcn_s_wait_tensorcnt((short)0);
      }
    }
    __syncthreads();
#else
    {
      float* dstb = (float*)lds_kv[pass & 1];
      const float* srcb = kvb + (size_t)pass * 32 * E_;
      for (int i = threadIdx.x * 4; i < 32 * E_; i += 256 * 4)
        *(float4*)&dstb[i] = *(const float4*)&srcb[i];
    }
    __syncthreads();
#endif
    // K and V projection: 32 s-rows x 64 head-cols; B-frags reused for 2 M-tiles.
    v8f aK0[4] = {}, aV0[4] = {}, aK1[4] = {}, aV1[4] = {};
    for (int kk = 0; kk < KSTEPS; ++kk) {
      Frag bK[4], bV[4];
#pragma unroll
      for (int t = 0; t < 4; ++t) {    // 16 x b128 loads batched in one clause
        size_t fo = ((size_t)((h * 4 + t) * KSTEPS + kk) * 32 + lane) * 2;
        bK[t].q[0] = wk4[fo]; bK[t].q[1] = wk4[fo + 1];
        bV[t].q[0] = wv4[fo]; bV[t].q[1] = wv4[fo + 1];
      }
      Frag a0 = make_afrag_f32(&cur[c16 * E_], kk, h2);        // rows 0..15
      Frag a1 = make_afrag_f32(&cur[(16 + c16) * E_], kk, h2); // rows 16..31
#pragma unroll
      for (int t = 0; t < 4; ++t) {    // 16 back-to-back WMMAs
        aK0[t] = __builtin_amdgcn_wmma_f32_16x16x32_f16(
            false, a0.v, false, bK[t].v, (short)0, aK0[t], false, false);
        aK1[t] = __builtin_amdgcn_wmma_f32_16x16x32_f16(
            false, a1.v, false, bK[t].v, (short)0, aK1[t], false, false);
        aV0[t] = __builtin_amdgcn_wmma_f32_16x16x32_f16(
            false, a0.v, false, bV[t].v, (short)0, aV0[t], false, false);
        aV1[t] = __builtin_amdgcn_wmma_f32_16x16x32_f16(
            false, a1.v, false, bV[t].v, (short)0, aV1[t], false, false);
      }
    }
    __syncthreads();   // all waves done reading `cur` before TDM refills it

    // logits for both M-tiles: q . (K+bias) over this wave's 64 columns.
    // C layout: lane holds col = t*16 + lane%16, rows j + 8*(lane/16).
    float lg0[8], lg1[8];
#pragma unroll
    for (int j = 0; j < 8; ++j) {
      float p0 = 0.f, p1 = 0.f;
#pragma unroll
      for (int t = 0; t < 4; ++t) {
        p0 += (aK0[t][j] + bkv[t]) * qv[t];
        p1 += (aK1[t][j] + bkv[t]) * qv[t];
      }
      lg0[j] = p0; lg1[j] = p1;
    }
#pragma unroll
    for (int j = 0; j < 8; ++j) {        // reduce across 16 cols (stays in half)
      lg0[j] += __shfl_xor(lg0[j], 1); lg1[j] += __shfl_xor(lg1[j], 1);
      lg0[j] += __shfl_xor(lg0[j], 2); lg1[j] += __shfl_xor(lg1[j], 2);
      lg0[j] += __shfl_xor(lg0[j], 4); lg1[j] += __shfl_xor(lg1[j], 4);
      lg0[j] += __shfl_xor(lg0[j], 8); lg1[j] += __shfl_xor(lg1[j], 8);
      lg0[j] *= 0.125f; lg1[j] *= 0.125f;   // 1/sqrt(64)
    }
    // online softmax over all 32 rows in one rescale step
    float lm = fmaxf(lg0[0], lg1[0]);
#pragma unroll
    for (int j = 1; j < 8; ++j) lm = fmaxf(lm, fmaxf(lg0[j], lg1[j]));
    lm = fmaxf(lm, __shfl_xor(lm, 16));
    float mnew = fmaxf(m, lm);
    float resc = __expf(m - mnew);
    float p0[8], p1[8], psum = 0.f;
#pragma unroll
    for (int j = 0; j < 8; ++j) {
      p0[j] = __expf(lg0[j] - mnew); p1[j] = __expf(lg1[j] - mnew);
      psum += p0[j] + p1[j];
    }
    lsum = lsum * resc + psum;
#pragma unroll
    for (int t = 0; t < 4; ++t) {
      float a = ctx[t] * resc;
#pragma unroll
      for (int j = 0; j < 8; ++j) {
        a += p0[j] * (aV0[t][j] + bvv[t]);
        a += p1[j] * (aV1[t][j] + bvv[t]);
      }
      ctx[t] = a;
    }
    m = mnew;
  }
  // combine the two row-halves, emit partial (unnormalized ctx, m, lsum)
#pragma unroll
  for (int t = 0; t < 4; ++t) ctx[t] += __shfl_xor(ctx[t], 16);
  float ltot = lsum + __shfl_xor(lsum, 16);
  float* po = part + ((size_t)(b * H_ + h) * NSB_ + sb) * 66;
  if (lane < 16) {
#pragma unroll
    for (int t = 0; t < 4; ++t) po[t * 16 + c16] = ctx[t];
  }
  if (lane == 0) { po[64] = m; po[65] = ltot; }
}

// ---------------- kernel 4: reduce S-block partials -> f16 context ----------
__global__ __launch_bounds__(64) void reduce_context_kernel(
    const float* __restrict__ part, _Float16* __restrict__ ctxh) {
  int bh = blockIdx.x, d = threadIdx.x;
  const float* base = part + (size_t)bh * NSB_ * 66;
  float M = -1e30f;
  for (int i = 0; i < NSB_; ++i) M = fmaxf(M, base[i * 66 + 64]);
  float s = 0.f, L = 0.f;
  for (int i = 0; i < NSB_; ++i) {
    float w = __expf(base[i * 66 + 64] - M);
    s += base[i * 66 + d] * w;
    L += base[i * 66 + 65] * w;
  }
  int b = bh / H_, h = bh % H_;
  ctxh[(size_t)b * E_ + h * 64 + d] = (_Float16)(s / L);
}

// ---------------- kernel 5: out = context @ Wo + bo -------------------------
__global__ __launch_bounds__(256) void oproj_kernel(
    const _Float16* __restrict__ ctxh, const unsigned* __restrict__ wop,
    const float* __restrict__ bo, float* __restrict__ out) {
  int lane = threadIdx.x & 31, w = threadIdx.x >> 5;
  int h2 = lane >> 4, c16 = lane & 15;
  const uint4* wp4 = (const uint4*)wop;
  v8f acc[4] = {};
  for (int kk = 0; kk < KSTEPS; ++kk) {
    Frag b[4];
#pragma unroll
    for (int t = 0; t < 4; ++t) {
      const uint4* p = wp4 + ((size_t)((w * 4 + t) * KSTEPS + kk) * 32 + lane) * 2;
      b[t].q[0] = p[0]; b[t].q[1] = p[1];
    }
    Frag a;   // A is already f16: 8+8 consecutive elements = 2 x b128
    a.q[0] = *(const uint4*)&ctxh[c16 * E_ + kk * 32 + h2 * 8];
    a.q[1] = *(const uint4*)&ctxh[c16 * E_ + kk * 32 + 16 + h2 * 8];
#pragma unroll
    for (int t = 0; t < 4; ++t)
      acc[t] = __builtin_amdgcn_wmma_f32_16x16x32_f16(
          false, a.v, false, b[t].v, (short)0, acc[t], false, false);
  }
#pragma unroll
  for (int t = 0; t < 4; ++t) {
    int c = (w * 4 + t) * 16 + c16;
    float bias = bo[c];
#pragma unroll
    for (int j = 0; j < 8; ++j) out[(j + 8 * h2) * E_ + c] = acc[t][j] + bias;
  }
}

// ---------------- host launcher ---------------------------------------------
extern "C" void kernel_launch(void* const* d_in, const int* in_sizes, int n_in,
                              void* d_out, int out_size, void* d_ws, size_t ws_size,
                              hipStream_t stream) {
  (void)in_sizes; (void)n_in; (void)out_size; (void)ws_size;
  const float* query = (const float*)d_in[0];
  const float* kv    = (const float*)d_in[1];
  const float* Wq    = (const float*)d_in[2];
  const float* bq    = (const float*)d_in[3];
  const float* Wk    = (const float*)d_in[4];
  const float* bk    = (const float*)d_in[5];
  const float* Wv    = (const float*)d_in[6];
  const float* bv    = (const float*)d_in[7];
  const float* Wo    = (const float*)d_in[8];
  const float* bo    = (const float*)d_in[9];

  char* w = (char*)d_ws;                       // ~2.7 MB used
  unsigned* wqp = (unsigned*)(w);              // 512 KB each packed matrix
  unsigned* wkp = (unsigned*)(w + (512u << 10));
  unsigned* wvp = (unsigned*)(w + (1024u << 10));
  unsigned* wop = (unsigned*)(w + (1536u << 10));
  float*    qws = (float*)(w + (2048u << 10));                 // 32 KB
  float*   part = (float*)(w + (2048u << 10) + 32768);         // 528 KB
  _Float16* ctxh =
      (_Float16*)(w + (2048u << 10) + 32768 + 540672);         // 16 KB

  pack_weights_kernel<<<2048, 256, 0, stream>>>(Wq, Wk, Wv, Wo, wqp, wkp, wvp, wop);
  qproj_kernel<<<1, 256, 0, stream>>>(query, wqp, bq, qws);
  attn_partial_kernel<<<B_ * NSB_, 256, 0, stream>>>(kv, qws, wkp, wvp, bk, bv, part);
  reduce_context_kernel<<<B_ * H_, 64, 0, stream>>>(part, ctxh);
  oproj_kernel<<<1, 256, 0, stream>>>(ctxh, wop, bo, (float*)d_out);
}
